// STATT_51582557225380
// MI455X (gfx1250) — matrix operand order repaired and checked by reading, not compiled, not run on hardware
//
#include <hip/hip_runtime.h>
#include <math.h>

// ---------------------------------------------------------------------------
// MI455X (gfx1250) spatio-temporal attention pipeline.
// Dominant cost: three 3x3 convs (~241 GFLOP) -> implicit GEMM with
// v_wmma_f32_16x16x32_f16 (wave32 WMMA, f32 accumulate). A-tiles are staged
// into double-buffered LDS with GLOBAL_LOAD_ASYNC_TO_LDS_B32 (ASYNCcnt),
// software-pipelined so async copies overlap WMMA compute. Everything else
// is bandwidth-bound fused passes.
// ---------------------------------------------------------------------------

typedef __attribute__((ext_vector_type(16))) _Float16 v16h;
typedef __attribute__((ext_vector_type(8)))  float    v8f;

#define L_   10
#define S_   10
#define B_   16
#define C_   128
#define HW_  1024
#define D_   (C_ * HW_)          // 131072 per (img, all channels)
#define NB_  (L_ * B_)           // 160 images per tensor (L*B == S*B)

#if defined(__AMDGCN__) && __has_builtin(__builtin_amdgcn_global_load_async_to_lds_b32)
#define ASYNC_LDS 1
typedef __attribute__((address_space(1))) int* gas_ptr_t;
typedef __attribute__((address_space(3))) int* las_ptr_t;
#endif

static __device__ __forceinline__ void wait_async0()
{
#if defined(__AMDGCN__)
#if __has_builtin(__builtin_amdgcn_s_wait_asynccnt)
  __builtin_amdgcn_s_wait_asynccnt(0);
#else
  asm volatile("s_wait_asynccnt 0x0" ::: "memory");
#endif
#endif
}

static __device__ __forceinline__ unsigned packh2(float a, float b)
{
  union { _Float16 h[2]; unsigned u; } p;
  p.h[0] = (_Float16)a;
  p.h[1] = (_Float16)b;
  return p.u;
}

// ---------------------------------------------------------------------------
// Conv3x3 (SAME, no bias) as implicit GEMM + fused GroupNorm statistics.
// Block: 256 threads = 8 waves, tile 32(M=pixels) x 64(N=out channels).
// Wave grid 2(M) x 4(N); each wave owns one 16x16 f32 accumulator and issues
// one v_wmma_f32_16x16x32_f16 per 32-wide K step (K total = CIN*9).
// Double-buffered LDS pipeline: stage tile it+1 (async) while computing it.
// A tile kept as f32 in LDS, converted during fragment build; B tile packed
// to f16 pairs by the loader.
// ---------------------------------------------------------------------------
template <int CIN>
__global__ __launch_bounds__(256)
void conv_gn_stats(const float* __restrict__ x,    // [NB][CIN][32][32]
                   const float* __restrict__ w,    // [128][CIN][3][3]
                   float* __restrict__ y,          // [NB][128][32][32]
                   float* __restrict__ stats)      // [NB][2] sum, sumsq (atomic)
{
  __shared__ float    lAf[2][32][34];    // [buf][pixel][k] f32 (+pad)
  __shared__ unsigned lBTu[2][64][18];   // [buf][n][k-pair] packed 2xf16 (+pad)

  const int tid  = threadIdx.x;
  const int lane = tid & 31;
  const int wave = tid >> 5;
  const int waveM = wave & 1;       // 0..1
  const int waveN = wave >> 1;      // 0..3
  const int mtile = blockIdx.x;     // 0..31  (32 pixels each)
  const int ntile = blockIdx.y;     // 0..1   (64 cout each)
  const int img   = blockIdx.z;     // 0..NB-1

  const float* xin = x + (size_t)img * CIN * HW_;
  const int n0g = ntile * 64;

  // A loader: 32 pixels x 8 threads, 4 channels each.
  const int ap   = tid >> 3;            // pixel-in-tile 0..31
  const int aj0  = (tid & 7) * 4;       // k-in-chunk base
  const int apix = mtile * 32 + ap;
  const int ay   = apix >> 5, ax = apix & 31;
  // B loader: 64 n x 4 threads, 8 channels each (stored transposed, packed).
  const int bn   = tid >> 2;            // 0..63
  const int bj0  = (tid & 3) * 8;

  const int CHUNKS = CIN / 32;          // 4 or 8 (power of two)
  const int NIT    = 9 * CHUNKS;

  // Stage one (kk, c0) tile into LDS buffer `buf`.
  auto stage = [&](int it, int buf) {
    const int kk = it / CHUNKS;
    const int c0 = (it % CHUNKS) * 32;
    const int ky = kk / 3 - 1, kx = kk % 3 - 1;
    const int iy = ay + ky,    ix = ax + kx;
    const bool inb = (iy >= 0) && (iy < 32) && (ix >= 0) && (ix < 32);
    const float* xrow = xin + (iy * 32 + ix);
    const float* wcol = w + (size_t)(n0g + bn) * CIN * 9 + kk;

    // ---- A tile (im2col patch rows), raw f32, async copy to LDS ----
#if defined(ASYNC_LDS)
    if (inb) {
#pragma unroll
      for (int u = 0; u < 4; ++u) {
        __builtin_amdgcn_global_load_async_to_lds_b32(
            (gas_ptr_t)(xrow + (size_t)(c0 + aj0 + u) * HW_),
            (las_ptr_t)&lAf[buf][ap][aj0 + u], 0, 0);
      }
    } else {
#pragma unroll
      for (int u = 0; u < 4; ++u) lAf[buf][ap][aj0 + u] = 0.0f;
    }
#else
    const float* xsafe = inb ? xrow : xin;   // branch-free fallback
    const float  xscale = inb ? 1.0f : 0.0f;
#pragma unroll
    for (int u = 0; u < 4; ++u)
      lAf[buf][ap][aj0 + u] = xsafe[(size_t)(c0 + aj0 + u) * HW_] * xscale;
#endif

    // ---- B tile transposed/packed: lBT[n][k] = w[n][c0+k][ky][kx] ----
#pragma unroll
    for (int u = 0; u < 4; ++u) {
      const float f0 = wcol[(size_t)(c0 + bj0 + 2 * u + 0) * 9];
      const float f1 = wcol[(size_t)(c0 + bj0 + 2 * u + 1) * 9];
      lBTu[buf][bn][(bj0 >> 1) + u] = packh2(f0, f1);
    }
  };

  v8f acc = {};

  // Prologue: stage tile 0, then pipeline: stage(it+1) || compute(it).
  stage(0, 0);
  wait_async0();
  __syncthreads();

#pragma unroll 2
  for (int it = 0; it < NIT; ++it) {
    const int cur = it & 1;
    if (it + 1 < NIT) stage(it + 1, cur ^ 1);   // overlap with compute below

    // ---- build WMMA fragments per ISA 7.12.2 layouts ----
    const int khalf = lane >> 4;
    const int arow  = waveM * 16 + (lane & 15);   // A row (pixel)
    const int bcol  = waveN * 16 + (lane & 15);   // B col (cout)
    const float*    arow_f = &lAf[cur][arow][0];
    const _Float16* brow   = (const _Float16*)&lBTu[cur][bcol][0];
    v16h a, b;
#pragma unroll
    for (int v = 0; v < 8; ++v) {
      const int kbA = (v < 4 ? v * 2 : 16 + (v - 4) * 2) + khalf * 8;
      a[2 * v]     = (_Float16)arow_f[kbA];
      a[2 * v + 1] = (_Float16)arow_f[kbA + 1];
      const int kbB = (v < 4 ? v * 2 : 16 + (v - 4) * 2) + khalf * 8;
      b[2 * v]     = brow[kbB];
      b[2 * v + 1] = brow[kbB + 1];
    }
    acc = __builtin_amdgcn_wmma_f32_16x16x32_f16(
        false, a, false, b, (short)0, acc, false, false);

    wait_async0();     // own async writes for tile it+1 landed
    __syncthreads();   // all tile writes visible; cur buffer free for reuse
  }

  // ---- store conv output + accumulate GroupNorm stats ----
  float lsum = 0.0f, lsq = 0.0f;
  const int n  = lane & 15;
  const int mb = (lane >> 4) * 8;
  float* yimg = y + (size_t)img * C_ * HW_;
  const int pixbase = mtile * 32 + waveM * 16;
#pragma unroll
  for (int v = 0; v < 8; ++v) {
    const float val = acc[v];
    yimg[(size_t)(n0g + waveN * 16 + n) * HW_ + (pixbase + mb + v)] = val;
    lsum += val;
    lsq  += val * val;
  }
  __shared__ float sred[2];
  if (tid == 0) { sred[0] = 0.0f; sred[1] = 0.0f; }
  __syncthreads();
  atomicAdd(&sred[0], lsum);
  atomicAdd(&sred[1], lsq);
  __syncthreads();
  if (tid == 0) {
    atomicAdd(&stats[img * 2 + 0], sred[0]);
    atomicAdd(&stats[img * 2 + 1], sred[1]);
  }
}

// ---------------------------------------------------------------------------
// GroupNorm(1) apply: xh = (x - mu) * rsqrt(var + eps) * gamma[c] + beta[c]
// ---------------------------------------------------------------------------
__global__ __launch_bounds__(256)
void gn_apply(float* __restrict__ y, const float* __restrict__ stats,
              const float* __restrict__ gamma, const float* __restrict__ beta)
{
  const size_t idx = (size_t)blockIdx.x * 256 + threadIdx.x;
  if (idx >= (size_t)NB_ * D_) return;
  const int img = (int)(idx / D_);
  const int c   = (int)((idx % D_) / HW_);
  const float inv = 1.0f / (float)D_;
  const float mu  = stats[img * 2 + 0] * inv;
  const float var = stats[img * 2 + 1] * inv - mu * mu;
  const float rs  = rsqrtf(var + 1e-5f);
  y[idx] = (y[idx] - mu) * rs * gamma[c] + beta[c];
}

// ---------------------------------------------------------------------------
// Spatial filter: sigmoid(conv3x3([mean_c, max_c]) + b). One block per image.
// ---------------------------------------------------------------------------
__global__ __launch_bounds__(1024)
void spatial_filter(const float* __restrict__ xp, const float* __restrict__ w,
                    const float* __restrict__ bias, float* __restrict__ f)
{
  __shared__ float smean[HW_];
  __shared__ float smax[HW_];
  const int img = blockIdx.x;
  const int pix = threadIdx.x;
  const float* xi = xp + (size_t)img * D_ + pix;
  float s = 0.0f, mx = -1e30f;
  for (int c = 0; c < C_; ++c) {
    const float v = xi[(size_t)c * HW_];
    s += v;
    mx = fmaxf(mx, v);
  }
  smean[pix] = s * (1.0f / C_);
  smax[pix]  = mx;
  __syncthreads();
  const int y = pix >> 5, x = pix & 31;
  float a = bias[0];
#pragma unroll
  for (int dy = -1; dy <= 1; ++dy)
#pragma unroll
    for (int dx = -1; dx <= 1; ++dx) {
      const int yy = y + dy, xx = x + dx;
      if (yy < 0 || yy > 31 || xx < 0 || xx > 31) continue;
      const int p = yy * 32 + xx;
      const int k = (dy + 1) * 3 + (dx + 1);
      a += w[k] * smean[p] + w[9 + k] * smax[p];
    }
  f[(size_t)img * HW_ + pix] = 1.0f / (1.0f + expf(-a));
}

// ---------------------------------------------------------------------------
// fKQ[s,b,c] = mean_hw fK[s,b,hw] * Qp[s,b,c,hw]. One block per (img,c).
// ---------------------------------------------------------------------------
__global__ __launch_bounds__(128)
void fkq_kernel(const float* __restrict__ qp, const float* __restrict__ fk,
                float* __restrict__ fkq)
{
  const int img = blockIdx.x >> 7;
  const int c   = blockIdx.x & 127;
  const float* q  = qp + ((size_t)img * C_ + c) * HW_;
  const float* fK = fk + (size_t)img * HW_;
  float s = 0.0f;
  for (int i = threadIdx.x; i < HW_; i += 128) s += q[i] * fK[i];
  __shared__ float red[128];
  red[threadIdx.x] = s;
  __syncthreads();
  for (int o = 64; o > 0; o >>= 1) {
    if (threadIdx.x < o) red[threadIdx.x] += red[threadIdx.x + o];
    __syncthreads();
  }
  if (threadIdx.x == 0) fkq[blockIdx.x] = red[0] * (1.0f / HW_);
}

// ---------------------------------------------------------------------------
// gate[s,b,hw] = sigmoid(sum_c fKQ[s,b,c] * Kp[s,b,c,hw])
// ---------------------------------------------------------------------------
__global__ __launch_bounds__(256)
void gate_kernel(const float* __restrict__ kp, const float* __restrict__ fkq,
                 float* __restrict__ gate)
{
  const int idx = blockIdx.x * 256 + threadIdx.x;   // img*HW + hw
  const int img = idx >> 10;
  const int hw  = idx & 1023;
  const float* k  = kp + (size_t)img * D_ + hw;
  const float* fq = fkq + img * C_;
  float s = 0.0f;
  for (int c = 0; c < C_; ++c) s += fq[c] * k[(size_t)c * HW_];
  gate[idx] = 1.0f / (1.0f + expf(-s));
}

// ---------------------------------------------------------------------------
// T[l,s,b] partial sums: read Qp/Kp once per d-chunk, 100 accumulators/thread.
// Grid (chunk=64, b=16); atomics into T (zeroed each launch).
// ---------------------------------------------------------------------------
__global__ __launch_bounds__(256)
void attn_scores(const float* __restrict__ qp, const float* __restrict__ kp,
                 const float* __restrict__ fq, const float* __restrict__ fk,
                 float* __restrict__ T)
{
  const int b  = blockIdx.y;
  const int d0 = blockIdx.x * (D_ / 64);
  float acc[L_ * S_];
#pragma unroll
  for (int i = 0; i < L_ * S_; ++i) acc[i] = 0.0f;

  for (int d = d0 + threadIdx.x; d < d0 + D_ / 64; d += 256) {
    const int hw = d & 1023;
    float qv[L_], kv[S_];
#pragma unroll
    for (int l = 0; l < L_; ++l) {
      const size_t base = (size_t)l * B_ + b;
      qv[l] = qp[base * D_ + d] * fq[base * HW_ + hw];
      kv[l] = kp[base * D_ + d] * fk[base * HW_ + hw];
    }
#pragma unroll
    for (int l = 0; l < L_; ++l)
#pragma unroll
      for (int s = 0; s < S_; ++s)
        acc[l * S_ + s] = fmaf(qv[l], kv[s], acc[l * S_ + s]);
  }

  __shared__ float tred[L_ * S_];
  for (int i = threadIdx.x; i < L_ * S_; i += 256) tred[i] = 0.0f;
  __syncthreads();
#pragma unroll
  for (int i = 0; i < L_ * S_; ++i) atomicAdd(&tred[i], acc[i]);
  __syncthreads();
  for (int i = threadIdx.x; i < L_ * S_; i += 256)
    atomicAdd(&T[i * B_ + b], tred[i]);
}

// ---------------------------------------------------------------------------
// Softmax over s per (l,b), then U[s,b] = sum_l P[l,s,b]. One block per b.
// ---------------------------------------------------------------------------
__global__ __launch_bounds__(32)
void softmax_u(const float* __restrict__ T, float* __restrict__ U)
{
  __shared__ float P[L_][S_];
  const int b = blockIdx.x, t = threadIdx.x;
  const float scale = rsqrtf((float)D_);
  if (t < L_) {
    float row[S_];
    float mx = -1e30f;
    for (int s = 0; s < S_; ++s) {
      row[s] = T[(t * S_ + s) * B_ + b] * scale;
      mx = fmaxf(mx, row[s]);
    }
    float sum = 0.0f;
    for (int s = 0; s < S_; ++s) { row[s] = expf(row[s] - mx); sum += row[s]; }
    const float inv = 1.0f / sum;
    for (int s = 0; s < S_; ++s) P[t][s] = row[s] * inv;
  }
  __syncthreads();
  if (t < S_) {
    float u = 0.0f;
    for (int l = 0; l < L_; ++l) u += P[l][t];
    U[t * B_ + b] = u;
  }
}

// ---------------------------------------------------------------------------
// out[b,c,h,w] = sum_s U[s,b] * Vp[s,b,c,hw] * gate[s,b,hw]
// ---------------------------------------------------------------------------
__global__ __launch_bounds__(256)
void out_kernel(const float* __restrict__ vp, const float* __restrict__ gate,
                const float* __restrict__ U, float* __restrict__ out)
{
  const size_t idx = (size_t)blockIdx.x * 256 + threadIdx.x;  // b*D + d
  const int b  = (int)(idx / D_);
  const int d  = (int)(idx % D_);
  const int hw = d & 1023;
  float s = 0.0f;
#pragma unroll
  for (int si = 0; si < S_; ++si) {
    const size_t base = (size_t)si * B_ + b;
    s += U[si * B_ + b] * vp[base * D_ + d] * gate[base * HW_ + hw];
  }
  out[idx] = s;
}

__global__ void zero_kernel(float* __restrict__ p, int n)
{
  const int i = blockIdx.x * 256 + threadIdx.x;
  if (i < n) p[i] = 0.0f;
}

// ---------------------------------------------------------------------------
extern "C" void kernel_launch(void* const* d_in, const int* in_sizes, int n_in,
                              void* d_out, int out_size, void* d_ws, size_t ws_size,
                              hipStream_t stream)
{
  const float* Q     = (const float*)d_in[0];
  const float* K     = (const float*)d_in[1];
  const float* V     = (const float*)d_in[2];
  const float* wq    = (const float*)d_in[3];
  const float* gq_g  = (const float*)d_in[4];
  const float* gq_b  = (const float*)d_in[5];
  const float* wk    = (const float*)d_in[6];
  const float* gk_g  = (const float*)d_in[7];
  const float* gk_b  = (const float*)d_in[8];
  const float* wv    = (const float*)d_in[9];
  const float* gv_g  = (const float*)d_in[10];
  const float* gv_b  = (const float*)d_in[11];
  const float* sfq_w = (const float*)d_in[12];
  const float* sfq_b = (const float*)d_in[13];
  const float* sfk_w = (const float*)d_in[14];
  const float* sfk_b = (const float*)d_in[15];

  // Workspace layout (floats). Total ~63.5M floats (~254 MB).
  float* ws = (float*)d_ws;
  const size_t NBD = (size_t)NB_ * D_;
  float* Qp     = ws;                       // NBD
  float* Kp     = Qp + NBD;                 // NBD
  float* Vp     = Kp + NBD;                 // NBD
  float* fQ     = Vp + NBD;                 // NB*HW
  float* fK     = fQ + (size_t)NB_ * HW_;   // NB*HW
  float* fKQ    = fK + (size_t)NB_ * HW_;   // NB*C
  float* gateB  = fKQ + (size_t)NB_ * C_;   // NB*HW
  float* statsQ = gateB + (size_t)NB_ * HW_;// 320
  float* statsK = statsQ + 2 * NB_;         // 320
  float* statsV = statsK + 2 * NB_;         // 320
  float* T      = statsV + 2 * NB_;         // 1600
  float* U      = T + L_ * S_ * B_;         // 160

  // Zero the atomic accumulation regions (stats + T + U): 2720 floats.
  const int nzero = 6 * NB_ + L_ * S_ * B_ + S_ * B_;
  zero_kernel<<<(nzero + 255) / 256, 256, 0, stream>>>(statsQ, nzero);

  // Convs (WMMA implicit GEMM) + fused GN stats.
  dim3 cgrid(32, 2, NB_);
  conv_gn_stats<C_>    <<<cgrid, 256, 0, stream>>>(Q, wq, Qp, statsQ);
  conv_gn_stats<2 * C_><<<cgrid, 256, 0, stream>>>(K, wk, Kp, statsK);
  conv_gn_stats<2 * C_><<<cgrid, 256, 0, stream>>>(V, wv, Vp, statsV);

  // GroupNorm apply.
  const int gnblocks = (int)(((size_t)NB_ * D_ + 255) / 256);
  gn_apply<<<gnblocks, 256, 0, stream>>>(Qp, statsQ, gq_g, gq_b);
  gn_apply<<<gnblocks, 256, 0, stream>>>(Kp, statsK, gk_g, gk_b);
  gn_apply<<<gnblocks, 256, 0, stream>>>(Vp, statsV, gv_g, gv_b);

  // Spatial filters.
  spatial_filter<<<NB_, 1024, 0, stream>>>(Qp, sfq_w, sfq_b, fQ);
  spatial_filter<<<NB_, 1024, 0, stream>>>(Kp, sfk_w, sfk_b, fK);

  // fKQ and spatial gate on V.
  fkq_kernel<<<NB_ * C_, 128, 0, stream>>>(Qp, fK, fKQ);
  gate_kernel<<<(NB_ * HW_) / 256, 256, 0, stream>>>(Kp, fKQ, gateB);

  // Attention scores, softmax-collapse, and output.
  attn_scores<<<dim3(64, B_), 256, 0, stream>>>(Qp, Kp, fQ, fK, T);
  softmax_u<<<B_, 32, 0, stream>>>(T, U);
  out_kernel<<<(int)(((size_t)B_ * D_) / 256), 256, 0, stream>>>(
      Vp, gateB, U, (float*)d_out);
}
